// FastPhaseAttention_25426206392416
// MI455X (gfx1250) — matrix operand order repaired and checked by reading, not compiled, not run on hardware
//
#include <hip/hip_runtime.h>
#include <cstdint>
#include <cstddef>

// ---------------------------------------------------------------------------
// FastPhaseAttention for MI455X (gfx1250, wave32, WMMA 16x16x32 f16,
// GLOBAL_LOAD_ASYNC_TO_LDS double-buffered pipeline).
//
// scores = clip( Qext . KextT , -10, 10 ) where the 544-dim extended feature
// packs [content_scale*q_norm (512) | w*cos/sin band stats (6) | zeros (26)].
// Flash-attention style online softmax; PV GEMM also via f16 WMMA.
// ---------------------------------------------------------------------------

typedef _Float16 v16h __attribute__((ext_vector_type(16)));
typedef float    v8f  __attribute__((ext_vector_type(8)));

#define B_  8
#define NQ  2048
#define NK  2048
#define DM  512
#define VD  512
#define DX  544      // 512 + 6 phase dims + 26 zero pad (multiple of 32)
#define KC  64       // keys per chunk

// ---- CDNA5 async copy: global -> LDS, tracked by ASYNCcnt -----------------
__device__ __forceinline__ void async_load_b128(uint32_t lds_addr, const void* gaddr) {
  asm volatile("global_load_async_to_lds_b128 %0, %1, off"
               :: "v"(lds_addr), "v"(gaddr) : "memory");
}
#if __has_builtin(__builtin_amdgcn_s_wait_asynccnt)
#define WAIT_ASYNC(n) __builtin_amdgcn_s_wait_asynccnt((unsigned short)(n))
#else
#define WAIT_ASYNC(n) asm volatile("s_wait_asynccnt %0" :: "i"(n) : "memory")
#endif
__device__ __forceinline__ uint32_t lds_addr_of(const void* p) {
  return (uint32_t)(uintptr_t)p;   // generic LDS-aperture addr[31:0] == LDS byte offset
}

// A operand (16x32 f16, contiguous layout):
// lanes 0-15: M=lane, K = kb+0..15 ; lanes 16-31: M=lane-16, K = kb+16..31
__device__ __forceinline__ v16h ld_a(const _Float16* row, int kb, int koff) {
  union { uint4 u[2]; v16h h; } t;
  t.u[0] = *(const uint4*)(row + kb + koff);
  t.u[1] = *(const uint4*)(row + kb + 8 + koff);
  return t.h;
}
// B operand (32x16 f16): lane n holds col N=n%16; lanes 0-15 K=kb+0..15,
// lanes 16-31 K=kb+16..31 (contiguous) -> caller pre-adds the +0/+16 offset.
__device__ __forceinline__ v16h ld_b(const _Float16* p) {
  union { uint4 u[2]; v16h h; } t;
  t.u[0] = *(const uint4*)(p);
  t.u[1] = *(const uint4*)(p + 8);
  return t.h;
}
__device__ __forceinline__ v8f vzero8() {
  v8f z = {0.f,0.f,0.f,0.f,0.f,0.f,0.f,0.f};
  return z;
}

// ---------------------------------------------------------------------------
// Kernel 1: per-row preprocess for Q and K -> extended f16 rows of length 544
// ---------------------------------------------------------------------------
__global__ __launch_bounds__(128) void fpa_rows(
    const float* __restrict__ qre, const float* __restrict__ qim,
    const float* __restrict__ kre, const float* __restrict__ kim,
    const float* __restrict__ wf,  const float* __restrict__ wm,
    const float* __restrict__ wsl, const float* __restrict__ csc,
    _Float16* __restrict__ qn, _Float16* __restrict__ kn)
{
  const int row = blockIdx.x;
  const int tid = threadIdx.x;
  const bool isQ = row < B_ * NQ;
  const int r = isQ ? row : row - B_ * NQ;
  const float* re = (isQ ? qre : kre) + (size_t)r * DM;
  const float* im = (isQ ? qim : kim) + (size_t)r * DM;
  _Float16* outr = (isQ ? qn : kn) + (size_t)r * DX;

  float4 r4 = ((const float4*)re)[tid];
  float4 i4 = ((const float4*)im)[tid];
  float reA[4] = {r4.x, r4.y, r4.z, r4.w};
  float imA[4] = {i4.x, i4.y, i4.z, i4.w};

  float s0 = 0.f, s1 = 0.f, s2 = 0.f, msq = 0.f;
#pragma unroll
  for (int j = 0; j < 4; ++j) {
    int idx = tid * 4 + j;
    float a = atan2f(imA[j], reA[j] + 1e-8f);
    if (idx < 170)      s0 += a;
    else if (idx < 341) s1 += a;
    else                s2 += a;
    msq += reA[j]*reA[j] + imA[j]*imA[j];
  }
#pragma unroll
  for (int off = 16; off > 0; off >>= 1) {
    s0  += __shfl_down(s0,  off);
    s1  += __shfl_down(s1,  off);
    s2  += __shfl_down(s2,  off);
    msq += __shfl_down(msq, off);
  }
  __shared__ float red[4][4];
  __shared__ float bc[7];
  const int wv = tid >> 5, ln = tid & 31;
  if (ln == 0) { red[wv][0]=s0; red[wv][1]=s1; red[wv][2]=s2; red[wv][3]=msq; }
  __syncthreads();
  if (tid == 0) {
    float t0=0.f, t1=0.f, t2=0.f, tm=0.f;
    for (int w = 0; w < 4; ++w) { t0+=red[w][0]; t1+=red[w][1]; t2+=red[w][2]; tm+=red[w][3]; }
    float mf = t0 * (1.0f/170.0f);
    float mm = t1 * (1.0f/171.0f);
    float ms = t2 * (1.0f/171.0f);
    float Wf = 1.f, Wm = 1.f, Ws = 1.f, sc = 1.f;
    if (isQ) { Wf = wf[0]; Wm = wm[0]; Ws = wsl[0]; sc = csc[0]; }  // fold w, scale into Q side
    bc[0] = Wf * cosf(mf); bc[1] = Wf * sinf(mf);
    bc[2] = Wm * cosf(mm); bc[3] = Wm * sinf(mm);
    bc[4] = Ws * cosf(ms); bc[5] = Ws * sinf(ms);
    bc[6] = sc / (sqrtf(tm) + 1e-8f);
  }
  __syncthreads();
  float invn = bc[6];
#pragma unroll
  for (int j = 0; j < 4; ++j) {
    float mag = sqrtf(reA[j]*reA[j] + imA[j]*imA[j]);
    outr[tid*4 + j] = (_Float16)(mag * invn);
  }
  if (tid < 32) outr[DM + tid] = (tid < 6) ? (_Float16)bc[tid] : (_Float16)0.0f;
}

// ---------------------------------------------------------------------------
// Kernel 1b: transpose V [b][k][v] f32 -> vT [b][v][k] f16 (LDS tiled)
// ---------------------------------------------------------------------------
__global__ __launch_bounds__(256) void fpa_vt(const float* __restrict__ v,
                                              _Float16* __restrict__ vT)
{
  __shared__ float tile[32][33];
  const int b  = blockIdx.z;
  const int k0 = blockIdx.x * 32;
  const int v0 = blockIdx.y * 32;
  const int tx = threadIdx.x, ty0 = threadIdx.y;   // block (32,8)
#pragma unroll
  for (int i = 0; i < 4; ++i) {
    int ty = ty0 + i * 8;
    tile[ty][tx] = v[((size_t)b*NK + k0 + ty) * VD + v0 + tx];
  }
  __syncthreads();
#pragma unroll
  for (int i = 0; i < 4; ++i) {
    int ty = ty0 + i * 8;
    vT[((size_t)b*VD + v0 + ty) * NK + k0 + tx] = (_Float16)tile[tx][ty];
  }
}

// ---------------------------------------------------------------------------
// Kernel 2: fused flash attention, 8 waves / 256 threads per block.
// Double-buffered K/V tiles fed by GLOBAL_LOAD_ASYNC_TO_LDS_B128.
// Waves 0-3 compute the score GEMM (16 keys each); all 8 waves split the
// 512 v-columns (64 each) for the PV GEMM.
// ---------------------------------------------------------------------------
constexpr int SMEM_BYTES = (16*DX + 2*KC*DX + 2*VD*KC) * 2  // q + 2*k + 2*v f16
                         + 16*KC*4                          // f32 score tile
                         + 16*KC*2                          // f16 P tile
                         + 48*4;                            // m/l/corr

// per-thread async op counts (256 threads)
#define KOPS 17   // KC*DX*2 / 16 / 256
#define VOPS 16   // VD*KC*2 / 16 / 256

__device__ __forceinline__ void issue_k(const _Float16* kn, const _Float16* kdst,
                                        int b, int k0, int tid) {
  const char* src = (const char*)(kn + ((size_t)b*NK + k0) * DX);
  uint32_t lbase = lds_addr_of(kdst);
#pragma unroll
  for (int i = 0; i < KOPS; ++i) {
    int idx = tid + i * 256;                 // uint4 index, KC*DX/8 = 4352 total
    async_load_b128(lbase + idx*16, src + (size_t)idx*16);
  }
}
__device__ __forceinline__ void issue_v(const _Float16* vT, const _Float16* vdst,
                                        int b, int k0, int tid) {
  const _Float16* src = vT + (size_t)b * VD * (size_t)NK + k0;
  uint32_t lbase = lds_addr_of(vdst);
#pragma unroll
  for (int i = 0; i < VOPS; ++i) {
    int idx = tid + i * 256;                 // 4096 total; vr=idx>>3, c=idx&7
    int vr = idx >> 3, c = idx & 7;
    async_load_b128(lbase + idx*16, src + (size_t)vr*NK + c*8);
  }
}

__global__ __launch_bounds__(256) void fpa_attn(
    const _Float16* __restrict__ qn, const _Float16* __restrict__ kn,
    const _Float16* __restrict__ vT, float* __restrict__ out)
{
  extern __shared__ char smem_raw[];
  _Float16* q_lds = (_Float16*)smem_raw;          // 16 x 544
  _Float16* kb0   = q_lds + 16*DX;                // 64 x 544, buffer 0
  _Float16* kb1   = kb0 + KC*DX;                  // 64 x 544, buffer 1
  _Float16* vb0   = kb1 + KC*DX;                  // 512 x 64, buffer 0
  _Float16* vb1   = vb0 + VD*KC;                  // 512 x 64, buffer 1
  float*    p_lds = (float*)(vb1 + VD*KC);        // 16 x 64 scores
  _Float16* p16   = (_Float16*)(p_lds + 16*KC);   // 16 x 64 exp probs
  float*    m_run = (float*)(p16 + 16*KC);
  float*    l_run = m_run + 16;
  float*    c_fac = l_run + 16;

  const int tid    = threadIdx.x;
  const int wave   = tid >> 5;
  const int lane   = tid & 31;
  const int lane16 = lane & 15;
  const int mhalf  = (lane < 16) ? 0 : 8;   // C/D row offset for this half-wave
  const int koff   = (lane < 16) ? 0 : 16;  // A operand K offset
  const int boff   = (lane < 16) ? 0 : 16;  // B operand K offset
  const int b      = blockIdx.y;
  const int q0     = blockIdx.x * 16;

  { // stage Q tile once (plain loads; small)
    const uint4* src = (const uint4*)(qn + ((size_t)b*NQ + q0) * DX);
    uint4* dst = (uint4*)q_lds;
    for (int i = tid; i < (16*DX)/8; i += 256) dst[i] = src[i];
  }
  if (tid < 16) { m_run[tid] = -1e30f; l_run[tid] = 0.0f; }

  // pipeline prologue: K(0), V(0), K(1) in flight (<= 50 ops/thread)
  issue_k(kn, kb0, b, 0, tid);
  issue_v(vT, vb0, b, 0, tid);
  issue_k(kn, kb1, b, KC, tid);

  v8f acc[4];
#pragma unroll
  for (int t = 0; t < 4; ++t) acc[t] = vzero8();

  const _Float16* arow = q_lds + lane16 * DX;
  const _Float16* prow = p16   + lane16 * KC;

  int cur = 0;
  for (int i = 0; i < NK/KC; ++i) {
    const _Float16* kc = cur ? kb1 : kb0;
    const _Float16* vc = cur ? vb1 : vb0;
    _Float16* knxt = cur ? kb0 : kb1;    // holds K(i) consumers' swap target
    _Float16* vnxt = cur ? vb0 : vb1;

    // K(i) complete: everything older than V(i)+K(i+1) (= 33 ops) retired
    if (i < NK/KC - 1) { WAIT_ASYNC(33); } else { WAIT_ASYNC(16); }
    __syncthreads();

    // ---- score GEMM (waves 0-3): S = Qext(16x544) . Kext(16 keys)^T ----
    if (wave < 4) {
      const _Float16* brow = kc + (wave*16 + lane16) * DX;
      v8f s = vzero8();
#pragma unroll
      for (int kk = 0; kk < DX; kk += 32) {
        v16h a  = ld_a(arow, kk, koff);
        v16h bb = ld_b(brow + kk + boff);
        s = __builtin_amdgcn_wmma_f32_16x16x32_f16(false, a, false, bb,
                                                   (short)0, s, false, false);
      }
#pragma unroll
      for (int r = 0; r < 8; ++r) {
        float x = s[r];
        x = fminf(10.0f, fmaxf(-10.0f, x));        // clip(-10, 10)
        p_lds[(r + mhalf)*KC + wave*16 + lane16] = x;
      }
    }

    // V(i) complete: only K(i+1) (17 ops) may remain
    if (i < NK/KC - 1) { WAIT_ASYNC(17); } else { WAIT_ASYNC(0); }
    __syncthreads();

    // ---- online softmax: 16 rows x 16 lanes, 4 cols each ----
    {
      const int row = tid >> 4, sub = tid & 15;
      const float* pr = p_lds + row*KC + sub*4;
      float mx = fmaxf(fmaxf(pr[0], pr[1]), fmaxf(pr[2], pr[3]));
#pragma unroll
      for (int off = 1; off < 16; off <<= 1) mx = fmaxf(mx, __shfl_xor(mx, off));
      float mold = m_run[row];
      float mnew = fmaxf(mold, mx);
      float corr = __expf(mold - mnew);
      float ls = 0.0f;
#pragma unroll
      for (int j = 0; j < 4; ++j) {
        float e = __expf(pr[j] - mnew);
        p16[row*KC + sub*4 + j] = (_Float16)e;
        ls += e;
      }
#pragma unroll
      for (int off = 1; off < 16; off <<= 1) ls += __shfl_xor(ls, off);
      if (sub == 0) {
        l_run[row] = l_run[row] * corr + ls;
        m_run[row] = mnew;
        c_fac[row] = corr;
      }
    }
    __syncthreads();

    // prefetch V(i+1) into the other buffer (its readers finished last iter)
    if (i + 1 < NK/KC) issue_v(vT, vnxt, b, (i + 1)*KC, tid);

    // ---- rescale accumulators, then PV GEMM (each wave: 64 v-cols) ----
    float cr[8];
#pragma unroll
    for (int r = 0; r < 8; ++r) cr[r] = c_fac[r + mhalf];
#pragma unroll
    for (int t = 0; t < 4; ++t)
#pragma unroll
      for (int r = 0; r < 8; ++r) acc[t][r] *= cr[r];

    v16h pa0 = ld_a(prow, 0,  koff);
    v16h pa1 = ld_a(prow, 32, koff);
#pragma unroll
    for (int t = 0; t < 4; ++t) {
      const _Float16* vrow = vc + (wave*64 + t*16 + lane16) * KC + boff;
      v16h b0 = ld_b(vrow);
      v16h b1 = ld_b(vrow + 32);
      acc[t] = __builtin_amdgcn_wmma_f32_16x16x32_f16(false, pa0, false, b0,
                                                      (short)0, acc[t], false, false);
      acc[t] = __builtin_amdgcn_wmma_f32_16x16x32_f16(false, pa1, false, b1,
                                                      (short)0, acc[t], false, false);
    }
    __syncthreads();   // all waves done reading kc/vc before they are refilled

    // prefetch K(i+2) into the other K buffer, not the one score just read
    if (i + 2 < NK/KC) issue_k(kn, knxt, b, (i + 2)*KC, tid);
    cur ^= 1;
  }

  // ---- epilogue: divide by l, write f32 output [b][q][v] ----
  float li[8];
#pragma unroll
  for (int r = 0; r < 8; ++r) li[r] = 1.0f / l_run[r + mhalf];
#pragma unroll
  for (int t = 0; t < 4; ++t) {
    int vcxy = wave*64 + t*16 + lane16;
#pragma unroll
    for (int r = 0; r < 8; ++r) {
      int m = r + mhalf;
      out[((size_t)b*NQ + q0 + m) * VD + vcxy] = acc[t][r] * li[r];
    }
  }
}

// ---------------------------------------------------------------------------
extern "C" void kernel_launch(void* const* d_in, const int* in_sizes, int n_in,
                              void* d_out, int out_size, void* d_ws, size_t ws_size,
                              hipStream_t stream) {
  (void)in_sizes; (void)n_in; (void)out_size; (void)ws_size;
  const float* qre = (const float*)d_in[0];
  const float* qim = (const float*)d_in[1];
  const float* kre = (const float*)d_in[2];
  const float* kim = (const float*)d_in[3];
  const float* val = (const float*)d_in[4];
  const float* wf  = (const float*)d_in[5];
  const float* wm  = (const float*)d_in[6];
  const float* wsl = (const float*)d_in[7];
  const float* csc = (const float*)d_in[8];
  float* out = (float*)d_out;

  _Float16* qn = (_Float16*)d_ws;                     // [B][NQ][544] f16
  _Float16* kn = qn + (size_t)B_ * NQ * DX;           // [B][NK][544] f16
  _Float16* vT = kn + (size_t)B_ * NK * DX;           // [B][VD][NK]  f16

  fpa_rows<<<dim3(B_ * (NQ + NK)), dim3(128), 0, stream>>>(
      qre, qim, kre, kim, wf, wm, wsl, csc, qn, kn);

  fpa_vt<<<dim3(NK/32, VD/32, B_), dim3(32, 8), 0, stream>>>(val, vT);

  hipFuncSetAttribute(reinterpret_cast<const void*>(fpa_attn),
                      hipFuncAttributeMaxDynamicSharedMemorySize, SMEM_BYTES);
  fpa_attn<<<dim3(NQ/16, B_), dim3(256), SMEM_BYTES, stream>>>(qn, kn, vT, out);
}
